// LocalAttention_50757923504844
// MI455X (gfx1250) — matrix operand order repaired and checked by reading, not compiled, not run on hardware
//
#include <hip/hip_runtime.h>
#include <hip/hip_bf16.h>

typedef __attribute__((ext_vector_type(16))) _Float16 v16h;
typedef __attribute__((ext_vector_type(8)))  _Float16 v8h;
typedef __attribute__((ext_vector_type(8)))  float    v8f;

#define S_LEN 4096
#define D_IN  1024
#define NH    16
#define HD    64
#define EMB   1024
#define WIN   128
#define VT_STRIDE 4352  // 4096 + 256 pad so banded vector loads stay in-bounds

union FragH { v16h v; v8h h[2]; };

static __device__ __forceinline__ int imin(int a, int b) { return a < b ? a : b; }
static __device__ __forceinline__ int imax(int a, int b) { return a > b ? a : b; }

static __device__ __forceinline__ v8f wmma16(v16h a, v16h b, v8f c) {
  return __builtin_amdgcn_wmma_f32_16x16x32_f16(false, a, false, b, (short)0, c,
                                                false, false);
}

// load a 16-half fragment as two contiguous 16-byte loads
static __device__ __forceinline__ v16h ldfrag(const _Float16* p) {
  FragH f;
  f.h[0] = *(const v8h*)(p);
  f.h[1] = *(const v8h*)(p + 8);
  return f.v;
}

// A-fragment element e -> k offset within a 32-wide chunk for lane l:
//   kb8 = (l>>4)*8 ; k = kb8 + e + (e>=8 ? 8 : 0)
// B-fragment element e -> k = (l>>4)*8 + e + (e>=8 ? 8 : 0) ; n = l&15

// ---------------------------------------------------------------------------
// Prep 1: weights (1024x1024 f32, row-major [k][n]) -> B-fragment-major f16.
// SW[nt(64)][c(32)][lane(32)][e(16)], grid=(32,64,4), block=32.
// ---------------------------------------------------------------------------
__global__ __launch_bounds__(32) void prep_weight_kernel(
    const float* __restrict__ Wq, const float* __restrict__ Wk,
    const float* __restrict__ Wv, const float* __restrict__ Wo,
    _Float16* __restrict__ SWq, _Float16* __restrict__ SWk,
    _Float16* __restrict__ SWv, _Float16* __restrict__ SWo)
{
  const int which = blockIdx.z;
  const float* W = (which == 0) ? Wq : (which == 1) ? Wk : (which == 2) ? Wv : Wo;
  _Float16* Out  = (which == 0) ? SWq : (which == 1) ? SWk : (which == 2) ? SWv : SWo;

  const int c  = blockIdx.x;
  const int nt = blockIdx.y;
  const int l  = threadIdx.x;
  const int n  = nt * 16 + (l & 15);
  const int k0 = c * 32 + (l >> 4) * 8;

  FragH f;
#pragma unroll
  for (int e = 0; e < 16; ++e) f.v[e] = (_Float16)W[(size_t)(k0 + e + (e >= 8 ? 8 : 0)) * 1024 + n];

  _Float16* dst = Out + (((size_t)nt * 32 + c) * 32 + l) * 16;
  *(v8h*)(dst)     = f.h[0];
  *(v8h*)(dst + 8) = f.h[1];
}

// ---------------------------------------------------------------------------
// Prep 2: activations (4096x1024 f32 row-major) -> A-fragment-major f16.
// As[mb(256)][c(32)][lane(32)][e(16)], grid=(32,256,3), block=32.
// ---------------------------------------------------------------------------
__global__ __launch_bounds__(32) void prep_act_kernel(
    const float* __restrict__ q_in, const float* __restrict__ k_in,
    const float* __restrict__ v_in,
    _Float16* __restrict__ Aq, _Float16* __restrict__ Ak,
    _Float16* __restrict__ Av)
{
  const int which = blockIdx.z;
  const float* X = (which == 0) ? q_in : (which == 1) ? k_in : v_in;
  _Float16* Out  = (which == 0) ? Aq : (which == 1) ? Ak : Av;

  const int c  = blockIdx.x;
  const int mb = blockIdx.y;
  const int l  = threadIdx.x;
  const int r  = mb * 16 + (l & 15);
  const int kb8 = (l >> 4) * 8;
  const float* src = X + (size_t)r * D_IN + c * 32 + kb8;

  FragH f;
#pragma unroll
  for (int e = 0; e < 8; ++e)  f.v[e] = (_Float16)src[e];
#pragma unroll
  for (int e = 8; e < 16; ++e) f.v[e] = (_Float16)src[e + 8];

  _Float16* dst = Out + (((size_t)mb * 32 + c) * 32 + l) * 16;
  *(v8h*)(dst)     = f.h[0];
  *(v8h*)(dst + 8) = f.h[1];
}

// ---------------------------------------------------------------------------
// Q/K/V projection GEMM, pure b128 + wmma inner loop.
// grid = (EMB/64, S/128, 3), block = 128 (4 waves); wave tile = 32(M) x 64(N).
// Q/K -> [h][s][64] f16 (Q scaled by 1/8); V -> transposed [h][d][VT_STRIDE].
// ---------------------------------------------------------------------------
__global__ __launch_bounds__(128) void proj_gemm_kernel(
    const _Float16* __restrict__ Aq, const _Float16* __restrict__ Ak,
    const _Float16* __restrict__ Av,
    const _Float16* __restrict__ SWq, const _Float16* __restrict__ SWk,
    const _Float16* __restrict__ SWv,
    _Float16* __restrict__ Qb, _Float16* __restrict__ Kb,
    _Float16* __restrict__ Vt)
{
  const int which = blockIdx.z;
  const _Float16* As = (which == 0) ? Aq : (which == 1) ? Ak : Av;
  const _Float16* Bs = (which == 0) ? SWq : (which == 1) ? SWk : SWv;

  const int nb   = blockIdx.x;
  const int mb   = blockIdx.y;
  const int wave = threadIdx.x >> 5;
  const int lane = threadIdx.x & 31;
  const int mblk0 = mb * 8 + wave * 2;  // two 16-row blocks per wave

  v8f acc[2][4] = {};

  for (int c = 0; c < D_IN / 32; ++c) {
    v16h a0 = ldfrag(As + (((size_t)(mblk0 + 0) * 32 + c) * 32 + lane) * 16);
    v16h a1 = ldfrag(As + (((size_t)(mblk0 + 1) * 32 + c) * 32 + lane) * 16);
#pragma unroll
    for (int nt = 0; nt < 4; ++nt) {
      v16h b = ldfrag(Bs + (((size_t)(nb * 4 + nt) * 32 + c) * 32 + lane) * 16);
      acc[0][nt] = wmma16(a0, b, acc[0][nt]);
      acc[1][nt] = wmma16(a1, b, acc[1][nt]);
    }
  }

  const int m8 = (lane >> 4) * 8;
  const int bn = lane & 15;
#pragma unroll
  for (int mi = 0; mi < 2; ++mi) {
    const int mrow0 = (mblk0 + mi) * 16 + m8;
    if (which == 0) {
#pragma unroll
      for (int nt = 0; nt < 4; ++nt) {
        int col = nb * 64 + nt * 16 + bn, h = col >> 6, d = col & 63;
#pragma unroll
        for (int i = 0; i < 8; ++i)
          Qb[((size_t)h * S_LEN + mrow0 + i) * HD + d] =
              (_Float16)(acc[mi][nt][i] * 0.125f);
      }
    } else if (which == 1) {
#pragma unroll
      for (int nt = 0; nt < 4; ++nt) {
        int col = nb * 64 + nt * 16 + bn, h = col >> 6, d = col & 63;
#pragma unroll
        for (int i = 0; i < 8; ++i)
          Kb[((size_t)h * S_LEN + mrow0 + i) * HD + d] = (_Float16)acc[mi][nt][i];
      }
    } else {
#pragma unroll
      for (int nt = 0; nt < 4; ++nt) {
        int col = nb * 64 + nt * 16 + bn, h = col >> 6, d = col & 63;
#pragma unroll
        for (int i = 0; i < 8; ++i)
          Vt[((size_t)h * HD + d) * VT_STRIDE + mrow0 + i] = (_Float16)acc[mi][nt][i];
      }
    }
  }
}

// ---------------------------------------------------------------------------
// Banded attention. grid = (S/16, NH), block = 32 (one wave).
// Output stored directly in A-fragment-major layout for the out projection.
// ---------------------------------------------------------------------------
__global__ __launch_bounds__(32) void attn_kernel(
    const _Float16* __restrict__ Qb, const _Float16* __restrict__ Kb,
    const _Float16* __restrict__ Vt, _Float16* __restrict__ AsO)
{
  const int qb   = blockIdx.x;
  const int h    = blockIdx.y;
  const int lane = threadIdx.x;

  const int kb_start = imax(0, qb - 8);
  const int kb_end   = imin(S_LEN / 16 - 1, qb + 8);
  const int nblk     = kb_end - kb_start + 1;  // <= 17
  const int nk       = nblk * 16;              // <= 272

  __shared__ float    sS[16][288];
  __shared__ _Float16 sP[16][288];

  {  // zero probs so padded PV columns contribute 0
    _Float16* p = &sP[0][0];
    for (int idx = lane; idx < 16 * 288; idx += 32) p[idx] = (_Float16)0.0f;
  }

  const int r   = lane & 15;
  const int kb8 = (lane >> 4) * 8;
  const int bn  = lane & 15;
  const int kh  = (lane >> 4) * 16;
  const int m8  = (lane >> 4) * 8;

  // Q fragments: two 32-wide K chunks, contiguous b128 loads
  const _Float16* qrow = Qb + ((size_t)h * S_LEN + qb * 16 + r) * HD;
  FragH aq0, aq1;
  aq0.h[0] = *(const v8h*)(qrow + kb8);
  aq0.h[1] = *(const v8h*)(qrow + kb8 + 16);
  aq1.h[0] = *(const v8h*)(qrow + 32 + kb8);
  aq1.h[1] = *(const v8h*)(qrow + 32 + kb8 + 16);

  // ---- scores = Q @ K^T, masked ----
  for (int blk = 0; blk < nblk; ++blk) {
    int key0 = (kb_start + blk) * 16;
    int krow = imin(key0 + bn, S_LEN - 1);
    const _Float16* kr = Kb + ((size_t)h * S_LEN + krow) * HD;
    FragH bk0, bk1;
    bk0.h[0] = *(const v8h*)(kr + kb8);
    bk0.h[1] = *(const v8h*)(kr + kb8 + 16);
    bk1.h[0] = *(const v8h*)(kr + 32 + kb8);
    bk1.h[1] = *(const v8h*)(kr + 32 + kb8 + 16);

    v8f c = {};
    c = wmma16(aq0.v, bk0.v, c);
    c = wmma16(aq1.v, bk1.v, c);
#pragma unroll
    for (int i = 0; i < 8; ++i) {
      int m  = m8 + i;
      int qi = qb * 16 + m;
      int ki = key0 + bn;
      bool valid = (ki < S_LEN) && (qi - ki <= WIN) && (ki - qi <= WIN);
      sS[m][blk * 16 + bn] = valid ? c[i] : -1e30f;
    }
  }
  __syncthreads();

  // ---- row softmax (lanes 0..15, one row each) ----
  if (lane < 16) {
    float mx = -1e30f;
    for (int c = 0; c < nk; ++c) mx = fmaxf(mx, sS[lane][c]);
    float sum = 0.0f;
    for (int c = 0; c < nk; ++c) {
      float e = __expf(sS[lane][c] - mx);
      sS[lane][c] = e;
      sum += e;
    }
    float inv = 1.0f / sum;
    for (int c = 0; c < nk; ++c) sP[lane][c] = (_Float16)(sS[lane][c] * inv);
  }
  __syncthreads();

  // ---- out = P @ V  (V fragments contiguous along keys via Vt) ----
  const int nkc = (nk + 31) / 32;
  v8f acc[4] = {};
  for (int c = 0; c < nkc; ++c) {
    FragH ap;
    ap.h[0] = *(const v8h*)(&sP[r][c * 32 + kb8]);
    ap.h[1] = *(const v8h*)(&sP[r][c * 32 + kb8 + 16]);
    int key0 = kb_start * 16 + c * 32;
#pragma unroll
    for (int nt = 0; nt < 4; ++nt) {
      const _Float16* vr =
          Vt + ((size_t)h * HD + nt * 16 + bn) * VT_STRIDE + key0 + kb8;
      FragH bv;
      bv.h[0] = *(const v8h*)(vr);
      bv.h[1] = *(const v8h*)(vr + 16);
      acc[nt] = wmma16(ap.v, bv.v, acc[nt]);
    }
  }

  // store 16x64 tile into A-fragment-major layout for the output projection:
  // AsO[mb(=qb)][c][lane][e]
#pragma unroll
  for (int nt = 0; nt < 4; ++nt) {
    int e   = (bn & 7) + ((nt & 1) << 3);
    int hf  = (bn >> 3) & 1;
    int cc  = h * 2 + (nt >> 1);
#pragma unroll
    for (int i = 0; i < 8; ++i) {
      int lp = hf * 16 + m8 + i;
      AsO[(((size_t)qb * 32 + cc) * 32 + lp) * 16 + e] = (_Float16)acc[nt][i];
    }
  }
}

// ---------------------------------------------------------------------------
// Output projection GEMM: AsO(f16, fragment-major) @ SWo -> fp32 out.
// grid = (D_IN/64, S/128), block = 128; wave tile 32x64.
// ---------------------------------------------------------------------------
__global__ __launch_bounds__(128) void out_proj_kernel(
    const _Float16* __restrict__ AsO, const _Float16* __restrict__ SWo,
    float* __restrict__ out)
{
  const int nb   = blockIdx.x;
  const int mb   = blockIdx.y;
  const int wave = threadIdx.x >> 5;
  const int lane = threadIdx.x & 31;
  const int mblk0 = mb * 8 + wave * 2;

  v8f acc[2][4] = {};

  for (int c = 0; c < EMB / 32; ++c) {
    v16h a0 = ldfrag(AsO + (((size_t)(mblk0 + 0) * 32 + c) * 32 + lane) * 16);
    v16h a1 = ldfrag(AsO + (((size_t)(mblk0 + 1) * 32 + c) * 32 + lane) * 16);
#pragma unroll
    for (int nt = 0; nt < 4; ++nt) {
      v16h b = ldfrag(SWo + (((size_t)(nb * 4 + nt) * 32 + c) * 32 + lane) * 16);
      acc[0][nt] = wmma16(a0, b, acc[0][nt]);
      acc[1][nt] = wmma16(a1, b, acc[1][nt]);
    }
  }

  const int m8 = (lane >> 4) * 8;
  const int bn = lane & 15;
#pragma unroll
  for (int mi = 0; mi < 2; ++mi) {
    const int mrow0 = (mblk0 + mi) * 16 + m8;
#pragma unroll
    for (int nt = 0; nt < 4; ++nt) {
      int col = nb * 64 + nt * 16 + bn;
#pragma unroll
      for (int i = 0; i < 8; ++i)
        out[(size_t)(mrow0 + i) * D_IN + col] = acc[mi][nt][i];
    }
  }
}

// ---------------------------------------------------------------------------
extern "C" void kernel_launch(void* const* d_in, const int* in_sizes, int n_in,
                              void* d_out, int out_size, void* d_ws,
                              size_t ws_size, hipStream_t stream) {
  const float* q  = (const float*)d_in[0];
  const float* k  = (const float*)d_in[1];
  const float* v  = (const float*)d_in[2];
  const float* Wq = (const float*)d_in[3];
  const float* Wk = (const float*)d_in[4];
  const float* Wv = (const float*)d_in[5];
  const float* Wo = (const float*)d_in[6];
  float* out = (float*)d_out;

  const size_t ACT_SZ = (size_t)S_LEN * D_IN;       // swizzled activation, halves
  const size_t W_SZ   = (size_t)1024 * 1024;        // swizzled weight, halves
  const size_t QK_SZ  = (size_t)NH * S_LEN * HD;    // [h][s][64], halves
  const size_t VT_SZ  = (size_t)NH * HD * VT_STRIDE;

  _Float16* p   = (_Float16*)d_ws;
  _Float16* Aq  = p;  p += ACT_SZ;
  _Float16* Ak  = p;  p += ACT_SZ;
  _Float16* Av  = p;  p += ACT_SZ;
  _Float16* SWq = p;  p += W_SZ;
  _Float16* SWk = p;  p += W_SZ;
  _Float16* SWv = p;  p += W_SZ;
  _Float16* SWo = p;  p += W_SZ;
  _Float16* Qb  = p;  p += QK_SZ;
  _Float16* Kb  = p;  p += QK_SZ;
  _Float16* Vt  = p;  p += VT_SZ;
  _Float16* AsO = p;  p += ACT_SZ;

  dim3 gw(32, 64, 4);
  prep_weight_kernel<<<gw, 32, 0, stream>>>(Wq, Wk, Wv, Wo, SWq, SWk, SWv, SWo);

  dim3 ga(32, 256, 3);
  prep_act_kernel<<<ga, 32, 0, stream>>>(q, k, v, Aq, Ak, Av);

  dim3 gproj(EMB / 64, S_LEN / 128, 3);
  proj_gemm_kernel<<<gproj, 128, 0, stream>>>(Aq, Ak, Av, SWq, SWk, SWv,
                                              Qb, Kb, Vt);

  dim3 gattn(S_LEN / 16, NH);
  attn_kernel<<<gattn, 32, 0, stream>>>(Qb, Kb, Vt, AsO);

  dim3 gout(D_IN / 64, S_LEN / 128);
  out_proj_kernel<<<gout, 128, 0, stream>>>(AsO, SWo, out);
}
